// CustomSelfAttention_5952824672316
// MI455X (gfx1250) — matrix operand order repaired and checked by reading, not compiled, not run on hardware
//
#include <hip/hip_runtime.h>
#include <math.h>

typedef __bf16 bf16_t;
typedef __attribute__((ext_vector_type(16))) __bf16 v16bf;
typedef __attribute__((ext_vector_type(8)))  __bf16 v8bf;
typedef __attribute__((ext_vector_type(8)))  float  v8f;
typedef __attribute__((ext_vector_type(4)))  float  v4f;

union ABf { v16bf v; v8bf h[2]; };

constexpr int Dd = 1024;   // feature dim
constexpr int Nn = 1024;   // sequence length
constexpr int Bb = 32;     // batch

// D = A(16x32 bf16) * B(32x16 bf16) + C(16x16 f32), wave32 CDNA5 WMMA
__device__ __forceinline__ v8f wmma_bf16(v16bf a, v16bf b, v8f c) {
  return __builtin_amdgcn_wmma_f32_16x16x32_bf16(false, a, false, b, (short)0, c,
                                                 false, false);
}

__device__ __forceinline__ v8bf load8bf(const bf16_t* p) { return *(const v8bf*)p; }

// load 8 consecutive f32 and convert to packed bf16 (v_cvt_pk_bf16_f32)
__device__ __forceinline__ v8bf cvt8(const float* p) {
  v4f a = *(const v4f*)p;
  v4f b = *(const v4f*)(p + 4);
  v8bf r;
  r[0] = (__bf16)a[0]; r[1] = (__bf16)a[1]; r[2] = (__bf16)a[2]; r[3] = (__bf16)a[3];
  r[4] = (__bf16)b[0]; r[5] = (__bf16)b[1]; r[6] = (__bf16)b[2]; r[7] = (__bf16)b[3];
  return r;
}

// A fragment (16 rows x 32 K, bf16): lane<16 -> row=lane, K {k0..k0+7, k0+16..k0+23};
// lane>=16 -> row=lane-16, K {k0+8.., k0+24..}. Row-major source, stride ld.
__device__ __forceinline__ v16bf loadA_bf16(const bf16_t* base, int ld, int k0, int lane) {
  int m = lane & 15, kb = (lane >> 4) * 8;
  const bf16_t* p = base + (size_t)m * ld + k0 + kb;
  ABf a; a.h[0] = load8bf(p); a.h[1] = load8bf(p + 16);
  return a.v;
}
// B fragment (32 K x 16 cols, bf16): lane<16 -> col=lane, K {k0..k0+7, k0+16..k0+23};
// lane>=16 -> col=lane-16, K {k0+8.., k0+24..}. Column n's K-run is contiguous at base+n*ld.
__device__ __forceinline__ v16bf loadB_bf16(const bf16_t* base, int ld, int k0, int lane) {
  int n = lane & 15, kb = (lane >> 4) * 8;
  const bf16_t* p = base + (size_t)n * ld + k0 + kb;
  ABf b; b.h[0] = load8bf(p); b.h[1] = load8bf(p + 16);
  return b.v;
}

// ---------------- Kernel 0: f32 -> bf16 bulk convert ----------------
// each thread converts 8 elements; n8 = n/8
__global__ void __launch_bounds__(256) cvt_kernel(const float* __restrict__ in,
                                                  bf16_t* __restrict__ out, int n8) {
  int i = blockIdx.x * 256 + threadIdx.x;
  if (i >= n8) return;
  *(v8bf*)(out + (size_t)i * 8) = cvt8(in + (size_t)i * 8);
}

// ---------------- Kernel 1: QKV projections ----------------
// grid (BN/128, D/16), block 128 (4 waves); each wave: 32 rows x 16 cols of q,k,v.
__global__ void __launch_bounds__(128) qkv_kernel(
    const bf16_t* __restrict__ xb,
    const bf16_t* __restrict__ wqb, const float* __restrict__ bq,
    const bf16_t* __restrict__ wkb, const float* __restrict__ bk,
    const bf16_t* __restrict__ wvb, const float* __restrict__ bv,
    bf16_t* __restrict__ q, bf16_t* __restrict__ k, bf16_t* __restrict__ vT) {
  int lane = threadIdx.x & 31, wave = threadIdx.x >> 5;
  int row0 = (blockIdx.x * 4 + wave) * 32;   // token rows (two 16-row tiles)
  int col0 = blockIdx.y * 16;                // output feature
  v8f cq0 = {}, ck0 = {}, cv0 = {};
  v8f cq1 = {}, ck1 = {}, cv1 = {};
  const bf16_t* xa0 = xb + (size_t)row0 * Dd;
  const bf16_t* xa1 = xa0 + (size_t)16 * Dd;
  const bf16_t* wqp = wqb + (size_t)col0 * Dd;
  const bf16_t* wkp = wkb + (size_t)col0 * Dd;
  const bf16_t* wvp = wvb + (size_t)col0 * Dd;
  for (int kk = 0; kk < Dd; kk += 32) {
    v16bf a0 = loadA_bf16(xa0, Dd, kk, lane);
    v16bf a1 = loadA_bf16(xa1, Dd, kk, lane);
    v16bf fq = loadB_bf16(wqp, Dd, kk, lane);   // W rows = B columns (x @ W^T)
    v16bf fk = loadB_bf16(wkp, Dd, kk, lane);
    v16bf fv = loadB_bf16(wvp, Dd, kk, lane);
    cq0 = wmma_bf16(a0, fq, cq0);  cq1 = wmma_bf16(a1, fq, cq1);
    ck0 = wmma_bf16(a0, fk, ck0);  ck1 = wmma_bf16(a1, fk, ck1);
    cv0 = wmma_bf16(a0, fv, cv0);  cv1 = wmma_bf16(a1, fv, cv1);
  }
  int n = lane & 15, half = lane >> 4;
  float biasq = bq[col0 + n], biask = bk[col0 + n], biasv = bv[col0 + n];
  int bidx = row0 / Nn;                       // 32 | N so tiles never cross batch
  for (int t = 0; t < 2; ++t) {
    const v8f& cq_ = t ? cq1 : cq0;
    const v8f& ck_ = t ? ck1 : ck0;
    const v8f& cv_ = t ? cv1 : cv0;
    int rbase = row0 + t * 16 + half * 8;
    v8bf pk;
    for (int r = 0; r < 8; ++r) {
      size_t off = (size_t)(rbase + r) * Dd + col0 + n;
      q[off] = (__bf16)(cq_[r] + biasq);
      k[off] = (__bf16)(ck_[r] + biask);
      pk[r]  = (__bf16)(cv_[r] + biasv);
    }
    // v stored transposed: vT[b][d][token], 8 consecutive tokens -> one b128 store
    *(v8bf*)(vT + (size_t)bidx * Dd * Nn + (size_t)(col0 + n) * Nn +
             (rbase - bidx * Nn)) = pk;
  }
}

// ---------------- Kernel 2: flash attention ----------------
// grid (B, N/16); block 128 = 4 waves. Each block: 16 queries x full D.
// Wave w owns output dims [w*256, w*256+256) -> 16 f32 accum fragments.
__global__ void __launch_bounds__(128) attn_kernel(
    const bf16_t* __restrict__ q, const bf16_t* __restrict__ kmat,
    const bf16_t* __restrict__ vT, const float* __restrict__ mask,
    bf16_t* __restrict__ ao) {
  __shared__ float Sp[4][16][32];                 // per-wave partial scores
  __shared__ __align__(16) bf16_t P[16][32];      // probabilities (bf16)
  __shared__ float mrow[16], lrow[16], srow[16], smask[32];
  int b = blockIdx.x, q0 = blockIdx.y * 16;
  int tid = threadIdx.x, lane = tid & 31, wave = tid >> 5;
  int d0 = wave * 256;
  int n = lane & 15, half = lane >> 4;
  v8f O[16];
  { v8f z = {}; for (int i = 0; i < 16; ++i) O[i] = z; }
  if (tid < 16) { mrow[tid] = -INFINITY; lrow[tid] = 0.f; }
  __syncthreads();
  const bf16_t* qbase = q + ((size_t)b * Nn + q0) * Dd;
  for (int kb0 = 0; kb0 < Nn; kb0 += 32) {
    // --- partial S = Q * K^T over this wave's D slice ---
    v8f s0 = {}, s1 = {};
    const bf16_t* k0base = kmat + ((size_t)b * Nn + kb0) * Dd;
    const bf16_t* k1base = k0base + (size_t)16 * Dd;
    for (int dd = d0; dd < d0 + 256; dd += 32) {
      v16bf a  = loadA_bf16(qbase, Dd, dd, lane);
      v16bf b0 = loadB_bf16(k0base, Dd, dd, lane);  // B col = key row (contig)
      v16bf b1 = loadB_bf16(k1base, Dd, dd, lane);
      s0 = wmma_bf16(a, b0, s0);
      s1 = wmma_bf16(a, b1, s1);
    }
    for (int r = 0; r < 8; ++r) {
      Sp[wave][half * 8 + r][n]      = s0[r];
      Sp[wave][half * 8 + r][16 + n] = s1[r];
    }
    if (wave == 0) smask[lane] = mask[(size_t)b * Nn + kb0 + lane];
    __syncthreads();
    // --- online softmax (threads 0..15, one query row each) ---
    if (tid < 16) {
      int r = tid;
      float mnew = mrow[r];
      for (int j = 0; j < 32; ++j) {
        float t = Sp[0][r][j] + Sp[1][r][j] + Sp[2][r][j] + Sp[3][r][j];
        Sp[0][r][j] = t;
        mnew = fmaxf(mnew, t);
      }
      float scale = __expf(mrow[r] - mnew);        // 0 on first block
      float ls = 0.f;
      for (int j = 0; j < 32; ++j) {
        float p = __expf(Sp[0][r][j] - mnew);
        ls += p;                                   // denominator: UNmasked
        P[r][j] = (__bf16)(p * smask[j]);          // numerator: masked
      }
      lrow[r] = lrow[r] * scale + ls;
      mrow[r] = mnew;
      srow[r] = scale;
    }
    __syncthreads();
    // --- rescale O, then O += P * V (K=32) over this wave's D slice ---
    float sc[8];
    for (int r = 0; r < 8; ++r) sc[r] = srow[half * 8 + r];
    ABf pa;
    { int m = lane & 15, akb = (lane >> 4) * 8;    // A-frag layout from LDS
      pa.h[0] = *(const v8bf*)&P[m][akb];
      pa.h[1] = *(const v8bf*)&P[m][akb + 16]; }
    const bf16_t* vb = vT + (size_t)b * Dd * Nn + kb0;  // vT[b][d][key]
    for (int i = 0; i < 16; ++i) {
      for (int r = 0; r < 8; ++r) O[i][r] *= sc[r];
      int dcol = d0 + i * 16;
      v16bf bb = loadB_bf16(vb + (size_t)dcol * Nn, Nn, 0, lane);
      O[i] = wmma_bf16(pa.v, bb, O[i]);
    }
    __syncthreads();
  }
  // --- finalize: divide by softmax denominator, store bf16 row-major ---
  float li[8];
  for (int r = 0; r < 8; ++r) li[r] = 1.f / lrow[half * 8 + r];
  for (int i = 0; i < 16; ++i) {
    int e = d0 + i * 16 + n;
    for (int r = 0; r < 8; ++r) {
      size_t trow = (size_t)b * Nn + q0 + half * 8 + r;
      ao[trow * Dd + e] = (__bf16)(O[i][r] * li[r]);
    }
  }
}

// ---------------- Kernel 3: output projection + bias + residual ----------------
// grid (BN/128, D/16), block 128; each wave: 32 rows x 16 cols
__global__ void __launch_bounds__(128) outproj_kernel(
    const bf16_t* __restrict__ ao, const bf16_t* __restrict__ wob,
    const float* __restrict__ bo, const float* __restrict__ x,
    float* __restrict__ y) {
  int lane = threadIdx.x & 31, wave = threadIdx.x >> 5;
  int row0 = (blockIdx.x * 4 + wave) * 32;
  int col0 = blockIdx.y * 16;
  v8f c0 = {}, c1 = {};
  const bf16_t* a0base = ao + (size_t)row0 * Dd;
  const bf16_t* a1base = a0base + (size_t)16 * Dd;
  const bf16_t* wbase = wob + (size_t)col0 * Dd;
  for (int kk = 0; kk < Dd; kk += 32) {
    v16bf a0 = loadA_bf16(a0base, Dd, kk, lane);
    v16bf a1 = loadA_bf16(a1base, Dd, kk, lane);
    v16bf bb = loadB_bf16(wbase, Dd, kk, lane);
    c0 = wmma_bf16(a0, bb, c0);
    c1 = wmma_bf16(a1, bb, c1);
  }
  int n = lane & 15, half = lane >> 4;
  float bias = bo[col0 + n];
  for (int t = 0; t < 2; ++t) {
    const v8f& c_ = t ? c1 : c0;
    int rbase = row0 + t * 16 + half * 8;
    for (int r = 0; r < 8; ++r) {
      size_t off = (size_t)(rbase + r) * Dd + col0 + n;
      y[off] = c_[r] + bias + x[off];
    }
  }
}

// ---------------- Kernel 4: in-place LayerNorm ----------------
// one block (256 threads) per row of 1024
__global__ void __launch_bounds__(256) ln_kernel(
    float* __restrict__ y, const float* __restrict__ gamma,
    const float* __restrict__ beta) {
  __shared__ float red[256];
  int tid = threadIdx.x;
  float* yr = y + (size_t)blockIdx.x * Dd;
  v4f v = *(const v4f*)(yr + tid * 4);
  red[tid] = v[0] + v[1] + v[2] + v[3];
  __syncthreads();
  for (int off = 128; off > 0; off >>= 1) {
    if (tid < off) red[tid] += red[tid + off];
    __syncthreads();
  }
  float mu = red[0] * (1.f / Dd);
  __syncthreads();
  float e0 = v[0] - mu, e1 = v[1] - mu, e2 = v[2] - mu, e3 = v[3] - mu;
  red[tid] = e0 * e0 + e1 * e1 + e2 * e2 + e3 * e3;
  __syncthreads();
  for (int off = 128; off > 0; off >>= 1) {
    if (tid < off) red[tid] += red[tid + off];
    __syncthreads();
  }
  float inv = rsqrtf(red[0] * (1.f / Dd) + 1e-5f);
  v4f g = *(const v4f*)(gamma + tid * 4);
  v4f be = *(const v4f*)(beta + tid * 4);
  v4f out;
  out[0] = e0 * inv * g[0] + be[0];
  out[1] = e1 * inv * g[1] + be[1];
  out[2] = e2 * inv * g[2] + be[2];
  out[3] = e3 * inv * g[3] + be[3];
  *(v4f*)(yr + tid * 4) = out;
}

extern "C" void kernel_launch(void* const* d_in, const int* in_sizes, int n_in,
                              void* d_out, int out_size, void* d_ws, size_t ws_size,
                              hipStream_t stream) {
  const float* x    = (const float*)d_in[0];
  const float* mask = (const float*)d_in[1];
  const float* wq = (const float*)d_in[2];  const float* bq = (const float*)d_in[3];
  const float* wk = (const float*)d_in[4];  const float* bk = (const float*)d_in[5];
  const float* wv = (const float*)d_in[6];  const float* bv = (const float*)d_in[7];
  const float* wo = (const float*)d_in[8];  const float* bo = (const float*)d_in[9];
  const float* gamma = (const float*)d_in[10];
  const float* beta  = (const float*)d_in[11];

  size_t elems = (size_t)Bb * Nn * Dd;       // 33.5M
  size_t welems = (size_t)Dd * Dd;           // 1M
  bf16_t* xb  = (bf16_t*)d_ws;               // 67 MB
  bf16_t* q   = xb + elems;                  // 67 MB each
  bf16_t* k   = q + elems;
  bf16_t* vT  = k + elems;
  bf16_t* ao  = vT + elems;
  bf16_t* wqb = ao + elems;                  // 2 MB each
  bf16_t* wkb = wqb + welems;
  bf16_t* wvb = wkb + welems;
  bf16_t* wob = wvb + welems;
  float* y = (float*)d_out;                  // d_out doubles as y scratch

  // bulk f32 -> bf16 conversions (do once; GEMM loops stay pure load+WMMA)
  int xn8 = (int)(elems / 8), wn8 = (int)(welems / 8);
  cvt_kernel<<<(xn8 + 255) / 256, 256, 0, stream>>>(x, xb, xn8);
  cvt_kernel<<<(wn8 + 255) / 256, 256, 0, stream>>>(wq, wqb, wn8);
  cvt_kernel<<<(wn8 + 255) / 256, 256, 0, stream>>>(wk, wkb, wn8);
  cvt_kernel<<<(wn8 + 255) / 256, 256, 0, stream>>>(wv, wvb, wn8);
  cvt_kernel<<<(wn8 + 255) / 256, 256, 0, stream>>>(wo, wob, wn8);

  dim3 gGemm(Bb * Nn / 128, Dd / 16);        // (256, 64)
  qkv_kernel<<<gGemm, 128, 0, stream>>>(xb, wqb, bq, wkb, bk, wvb, bv, q, k, vT);
  attn_kernel<<<dim3(Bb, Nn / 16), 128, 0, stream>>>(q, k, vT, mask, ao);
  outproj_kernel<<<gGemm, 128, 0, stream>>>(ao, wob, bo, x, y);
  ln_kernel<<<Bb * Nn, 256, 0, stream>>>(y, gamma, beta);
}
